// HybridPhysicsGNN_85718957294177
// MI455X (gfx1250) — matrix operand order repaired and checked by reading, compiled, not run on hardware
//
#include <hip/hip_runtime.h>

// ---------------------------------------------------------------------------
// HybridPhysicsGNN for MI455X (gfx1250, wave32, WMMA)
// N=50000 nodes, E=800000 edges, G=16, D=64, H=4, L=4, IN=7
// ---------------------------------------------------------------------------

#define NN     50000
#define EE     800000
#define GG     16
#define DD     64
#define HH     4
#define LL     4
#define ININ   7
#define NEGS   0.2f
#define LNEPS  1e-5f

typedef __bf16 bf16;
typedef __attribute__((ext_vector_type(16))) __bf16 bf16x16;
typedef __attribute__((ext_vector_type(8)))  __bf16 bf16x8;
typedef __attribute__((ext_vector_type(8)))  float  f32x8;

// ----------------------------- small helpers -------------------------------

__device__ __forceinline__ float lrelu(float v) { return v > 0.f ? v : NEGS * v; }

__device__ __forceinline__ void atomicMaxF(float* addr, float val) {
    unsigned int* ua = (unsigned int*)addr;
    unsigned int cur = *ua;
    while (__uint_as_float(cur) < val) {
        unsigned int old = atomicCAS(ua, cur, __float_as_uint(val));
        if (old == cur) break;
        cur = old;
    }
}

__global__ void fill_f32_kernel(float* __restrict__ p, float v, long n) {
    long t = (long)blockIdx.x * blockDim.x + threadIdx.x;
    if (t < n) p[t] = v;
}

// ---------------------- edge_attr mean (for self loops) --------------------

__global__ void ea_sum_kernel(const float* __restrict__ ea, float* __restrict__ sums, int E) {
    long stride = (long)gridDim.x * blockDim.x;
    long t = (long)blockIdx.x * blockDim.x + threadIdx.x;
    float s0 = 0.f, s1 = 0.f, s2 = 0.f;
    for (long e = t; e < E; e += stride) {
        s0 += ea[e * 3 + 0]; s1 += ea[e * 3 + 1]; s2 += ea[e * 3 + 2];
    }
    for (int off = 16; off; off >>= 1) {
        s0 += __shfl_xor(s0, off); s1 += __shfl_xor(s1, off); s2 += __shfl_xor(s2, off);
    }
    if ((threadIdx.x & 31) == 0) {
        atomicAdd(&sums[0], s0); atomicAdd(&sums[1], s1); atomicAdd(&sums[2], s2);
    }
}

__global__ void ea_mean_kernel(float* __restrict__ sums, float invE) {
    if (threadIdx.x < 3) sums[threadIdx.x] *= invE;
}

// ------------------------------- encoder -----------------------------------

__global__ void encoder_kernel(const float* __restrict__ x, const float* __restrict__ W,
                               const float* __restrict__ b, float* __restrict__ h,
                               bf16* __restrict__ hb, int N) {
    long t = (long)blockIdx.x * blockDim.x + threadIdx.x;
    if (t >= (long)N * DD) return;
    int n = (int)(t >> 6), d = (int)(t & 63);
    float z = b[d];
#pragma unroll
    for (int k = 0; k < ININ; ++k) z += x[(long)n * ININ + k] * W[k * DD + d];
    h[t] = z;
    hb[t] = (bf16)z;
}

// ------------------- weight repack: f32 [K,Nc] -> WMMA B fragments ---------
// Packed layout: P[((tn*KS + ks)*32 + lane)*16 + j] = B[k, n],
//   k = ks*32 + (lane>>4)*8 + (j&7) + (j>>3)*16, n = tn*16 + (lane&15), KS = K/32.
// Each lane of the GEMM then loads its 16 halves as two contiguous b128s.

__global__ void repack_b_kernel(const float* __restrict__ B, bf16* __restrict__ P,
                                int K, int Nc, int KS) {
    long t = (long)blockIdx.x * blockDim.x + threadIdx.x;
    if (t >= (long)K * Nc) return;
    int j    = (int)(t & 15);
    int lane = (int)((t >> 4) & 31);
    long rest = t >> 9;
    int ks = (int)(rest % KS);
    int tn = (int)(rest / KS);
    int k = ks * 32 + (lane >> 4) * 8 + (j & 7) + (j >> 3) * 16;
    int n = tn * 16 + (lane & 15);
    P[t] = (bf16)B[(long)k * Nc + n];
}

// ----------------------- bf16 WMMA GEMM  out = act(A@B + bias) -------------
// A [M,K] bf16 row-major, PB packed B fragments.
// OMODE 0: store f32 to C.  OMODE 1: store bf16 to Cb.
// One wave32 per 16x16 tile, K fully unrolled; guards are wave-uniform so
// EXEC is all-1s at every WMMA.

template <int K, int ACT, int OMODE>
__global__ void gemm_bf16_wmma_kernel(const bf16* __restrict__ A, const bf16* __restrict__ PB,
                                      const float* __restrict__ bias, float* __restrict__ C,
                                      bf16* __restrict__ Cb, int M, int Nc) {
    constexpr int KS = K / 32;
    int waveId = threadIdx.x >> 5;
    int lane   = threadIdx.x & 31;
    int tilesN = Nc >> 4;
    long tile  = (long)blockIdx.x * 8 + waveId;
    long numTiles = (long)(M >> 4) * tilesN;
    if (tile >= numTiles) return;                 // wave-uniform
    int tm = (int)(tile / tilesN);
    int tn = (int)(tile % tilesN);
    int hi = lane >> 4;                           // 0: lanes 0-15, 1: lanes 16-31
    int row = tm * 16 + (lane & 15);              // A row owned by this lane
    int col = tn * 16 + (lane & 15);              // C column owned by this lane

    f32x8 acc = {};
#pragma unroll
    for (int ks = 0; ks < KS; ++ks) {
        // A 16x32 bf16 fragment: elems 0..7 -> K=ka..ka+7, elems 8..15 -> K=ka+16..ka+23
        const bf16* ap = A + (long)row * K + ks * 32 + hi * 8;
        bf16x8 a0 = *(const bf16x8*)(ap);
        bf16x8 a1 = *(const bf16x8*)(ap + 16);
        // B fragment: 16 contiguous halves from the packed buffer
        const bf16* bp = PB + (((long)tn * KS + ks) << 9) + (lane << 4);
        bf16x8 b0 = *(const bf16x8*)(bp);
        bf16x8 b1 = *(const bf16x8*)(bp + 8);
        bf16x16 a, bm;
#pragma unroll
        for (int i = 0; i < 8; ++i) { a[i] = a0[i]; a[8 + i] = a1[i]; bm[i] = b0[i]; bm[8 + i] = b1[i]; }
        acc = __builtin_amdgcn_wmma_f32_16x16x32_bf16(false, a, false, bm,
                                                      (short)0, acc, false, false);
    }
    // C/D layout: VGPR g -> row = hi*8 + g, col = lane&15
    int rbase = tm * 16 + hi * 8;
    float bia = bias[col];
#pragma unroll
    for (int g = 0; g < 8; ++g) {
        float v = acc[g] + bia;
        if (ACT == 1) v = lrelu(v);
        long idx = (long)(rbase + g) * Nc + col;
        if (OMODE == 0) C[idx]  = v;
        else            Cb[idx] = (bf16)v;
    }
}

// --------------------------- GATv2 edge passes -----------------------------
// One wave32 per edge; lane owns 8 contiguous channels (head = lane>>3),
// so each gathered row is read with two b128 loads per operand.

__device__ __forceinline__ float edge_term4(float4 xl, float4 xr, float4 w0, float4 w1,
                                            float4 w2, float4 at, float a0, float a1, float a2) {
    float p;
    p  = lrelu(xl.x + xr.x + a0 * w0.x + a1 * w1.x + a2 * w2.x) * at.x;
    p += lrelu(xl.y + xr.y + a0 * w0.y + a1 * w1.y + a2 * w2.y) * at.y;
    p += lrelu(xl.z + xr.z + a0 * w0.z + a1 * w1.z + a2 * w2.z) * at.z;
    p += lrelu(xl.w + xr.w + a0 * w0.w + a1 * w1.w + a2 * w2.w) * at.w;
    return p;
}

__global__ void edge_logits_kernel(const float* __restrict__ xl, const float* __restrict__ xr,
                                   const float* __restrict__ ea, const int* __restrict__ ei,
                                   const float* __restrict__ eamean, const float* __restrict__ We,
                                   const float* __restrict__ attw, float* __restrict__ logits,
                                   float* __restrict__ mx, int E, int N) {
    int wave = threadIdx.x >> 5, lane = threadIdx.x & 31;
    long e = (long)blockIdx.x * 8 + wave;
    long Etot = (long)E + N;
    if (e >= Etot) return;
    int s, d; float a0, a1, a2;
    if (e < E) {
        s = ei[e]; d = ei[(long)E + e];
        a0 = ea[e * 3 + 0]; a1 = ea[e * 3 + 1]; a2 = ea[e * 3 + 2];
    } else {
        s = d = (int)(e - E);
        a0 = eamean[0]; a1 = eamean[1]; a2 = eamean[2];
    }
    int c0 = lane * 8;                 // channel base owned by this lane
    int hh = lane >> 3;                // head index
    const float4* xla = (const float4*)(xl + (long)s * 256 + c0);
    const float4* xra = (const float4*)(xr + (long)d * 256 + c0);
    const float4* We0 = (const float4*)(We + c0);
    const float4* We1 = (const float4*)(We + 256 + c0);
    const float4* We2 = (const float4*)(We + 512 + c0);
    const float4* aw  = (const float4*)(attw + c0);
    float p = edge_term4(xla[0], xra[0], We0[0], We1[0], We2[0], aw[0], a0, a1, a2)
            + edge_term4(xla[1], xra[1], We0[1], We1[1], We2[1], aw[1], a0, a1, a2);
    // segmented reduction over the 8 lanes of this head
    p += __shfl_xor(p, 4);
    p += __shfl_xor(p, 2);
    p += __shfl_xor(p, 1);
    if ((lane & 7) == 0) {
        logits[e * 4 + hh] = p;
        atomicMaxF(&mx[(long)d * 4 + hh], p);
    }
}

__global__ void edge_exp_kernel(float* __restrict__ logits, const float* __restrict__ mx,
                                float* __restrict__ den, const int* __restrict__ ei,
                                int E, int N) {
    long t = (long)blockIdx.x * blockDim.x + threadIdx.x;
    long n4 = ((long)E + N) * 4;
    if (t >= n4) return;
    long e = t >> 2; int hh = (int)(t & 3);
    int d = (e < E) ? ei[(long)E + e] : (int)(e - E);
    float ex = __expf(logits[t] - mx[(long)d * 4 + hh]);
    logits[t] = ex;                       // overwrite logits with exp values
    atomicAdd(&den[(long)d * 4 + hh], ex);
}

__global__ void edge_agg_kernel(const float* __restrict__ exv, const float* __restrict__ den,
                                const float* __restrict__ xl, const int* __restrict__ ei,
                                float* __restrict__ agg, int E, int N) {
    int wave = threadIdx.x >> 5, lane = threadIdx.x & 31;
    long e = (long)blockIdx.x * 8 + wave;
    long Etot = (long)E + N;
    if (e >= Etot) return;
    int s, d;
    if (e < E) { s = ei[e]; d = ei[(long)E + e]; }
    else       { s = d = (int)(e - E); }
    int c0 = lane * 8;
    int hh = lane >> 3;
    float alpha = exv[e * 4 + hh] / den[(long)d * 4 + hh];
    const float4* xla = (const float4*)(xl + (long)s * 256 + c0);
    float4 v0 = xla[0], v1 = xla[1];
    float* dst = agg + (long)d * 256 + c0;
    atomicAdd(dst + 0, alpha * v0.x);
    atomicAdd(dst + 1, alpha * v0.y);
    atomicAdd(dst + 2, alpha * v0.z);
    atomicAdd(dst + 3, alpha * v0.w);
    atomicAdd(dst + 4, alpha * v1.x);
    atomicAdd(dst + 5, alpha * v1.y);
    atomicAdd(dst + 6, alpha * v1.z);
    atomicAdd(dst + 7, alpha * v1.w);
}

// --------------- residual + LayerNorm (wave per node, dual output) ---------

__device__ __forceinline__ void wave_ln_store(float* __restrict__ h, bf16* __restrict__ hb,
                                              float t0, float t1,
                                              const float* __restrict__ w,
                                              const float* __restrict__ b,
                                              long n, int lane) {
    float sum = t0 + t1, sq = t0 * t0 + t1 * t1;
    for (int off = 16; off; off >>= 1) { sum += __shfl_xor(sum, off); sq += __shfl_xor(sq, off); }
    float m = sum * (1.f / 64.f);
    float v = sq * (1.f / 64.f) - m * m;
    float r = rsqrtf(v + LNEPS);
    int d0 = lane, d1 = lane + 32;
    float o0 = (t0 - m) * r * w[d0] + b[d0];
    float o1 = (t1 - m) * r * w[d1] + b[d1];
    h[n * 64 + d0] = o0;  hb[n * 64 + d0] = (bf16)o0;
    h[n * 64 + d1] = o1;  hb[n * 64 + d1] = (bf16)o1;
}

__global__ void gat_ln_kernel(float* __restrict__ h, bf16* __restrict__ hb,
                              const float* __restrict__ agg, const float* __restrict__ gatb,
                              const float* __restrict__ w, const float* __restrict__ b, int N) {
    int wave = threadIdx.x >> 5, lane = threadIdx.x & 31;
    long n = (long)blockIdx.x * 8 + wave;
    if (n >= N) return;
    float t[2];
#pragma unroll
    for (int i = 0; i < 2; ++i) {
        int d = lane + i * 32;
        const float* a = agg + n * 256 + d;
        float g = 0.25f * (a[0] + a[64] + a[128] + a[192]) + gatb[d];
        t[i] = h[n * 64 + d] + g;
    }
    wave_ln_store(h, hb, t[0], t[1], w, b, n, lane);
}

__global__ void add_ln_kernel(float* __restrict__ h, bf16* __restrict__ hb,
                              const float* __restrict__ add, const float* __restrict__ w,
                              const float* __restrict__ b, int N) {
    int wave = threadIdx.x >> 5, lane = threadIdx.x & 31;
    long n = (long)blockIdx.x * 8 + wave;
    if (n >= N) return;
    float t0 = h[n * 64 + lane]      + add[n * 64 + lane];
    float t1 = h[n * 64 + lane + 32] + add[n * 64 + lane + 32];
    wave_ln_store(h, hb, t0, t1, w, b, n, lane);
}

// ------------------------- global context layer ----------------------------

__global__ void seg_reduce_kernel(const float* __restrict__ lat, const int* __restrict__ batch,
                                  float* __restrict__ gsum, float* __restrict__ gmax,
                                  float* __restrict__ cnt, int N) {
    long t = (long)blockIdx.x * blockDim.x + threadIdx.x;
    if (t >= (long)N * 64) return;
    int n = (int)(t >> 6), d = (int)(t & 63);
    int g = batch[n];
    float v = lat[t];
    atomicAdd(&gsum[g * 64 + d], v);
    atomicMaxF(&gmax[g * 64 + d], v);
    if (d == 0) atomicAdd(&cnt[g], 1.f);
}

__global__ void gate_kernel(const float* __restrict__ gsum, const float* __restrict__ gmax,
                            const float* __restrict__ cnt, const float* __restrict__ W,
                            const float* __restrict__ b, float* __restrict__ gate) {
    int g = threadIdx.x >> 6, d = threadIdx.x & 63;   // 1024 threads = 16x64
    float c = cnt[g]; c = c > 1.f ? c : 1.f;
    float z = b[d];
#pragma unroll 4
    for (int k = 0; k < 64; ++k) {
        z += (gsum[g * 64 + k] / c) * W[k * 64 + d];
        z += gmax[g * 64 + k] * W[(64 + k) * 64 + d];
    }
    gate[g * 64 + d] = 1.f / (1.f + __expf(-z));
}

__global__ void apply_gate_kernel(float* __restrict__ h, bf16* __restrict__ hb,
                                  const float* __restrict__ gate,
                                  const int* __restrict__ batch, int N) {
    long t = (long)blockIdx.x * blockDim.x + threadIdx.x;
    if (t >= (long)N * 64) return;
    int n = (int)(t >> 6), d = (int)(t & 63);
    float v = h[t] * gate[batch[n] * 64 + d];
    h[t] = v;
    hb[t] = (bf16)v;
}

// ------------------------------- decoder tail ------------------------------

__global__ void dec2_kernel(const float* __restrict__ tbuf, const float* __restrict__ W2,
                            const float* __restrict__ b2, float* __restrict__ out, int N) {
    long t = (long)blockIdx.x * blockDim.x + threadIdx.x;
    if (t >= (long)N * 2) return;
    int n = (int)(t >> 1), j = (int)(t & 1);
    float z = b2[j];
#pragma unroll 4
    for (int k = 0; k < 64; ++k) z += tbuf[(long)n * 64 + k] * W2[k * 2 + j];
    out[t] = z;
}

// ============================== host launcher ==============================

extern "C" void kernel_launch(void* const* d_in, const int* in_sizes, int n_in,
                              void* d_out, int out_size, void* d_ws, size_t ws_size,
                              hipStream_t stream) {
    (void)in_sizes; (void)n_in; (void)out_size; (void)ws_size;

    // -------- inputs (setup_inputs dict order) --------
    const float* x        = (const float*)d_in[0];
    const float* ea       = (const float*)d_in[1];
    const int*   ei       = (const int*)  d_in[2];   // [2,E]
    const int*   batch    = (const int*)  d_in[3];
    const float* enc_W    = (const float*)d_in[4];
    const float* enc_b    = (const float*)d_in[5];
    const float* Wl       = (const float*)d_in[6];
    const float* bl       = (const float*)d_in[7];
    const float* Wr       = (const float*)d_in[8];
    const float* br       = (const float*)d_in[9];
    const float* We       = (const float*)d_in[10];
    const float* att_w    = (const float*)d_in[11];
    const float* gat_b    = (const float*)d_in[12];
    const float* ln1_w    = (const float*)d_in[13];
    const float* ln1_b    = (const float*)d_in[14];
    const float* ln2_w    = (const float*)d_in[15];
    const float* ln2_b    = (const float*)d_in[16];
    const float* ffn_W1   = (const float*)d_in[17];
    const float* ffn_b1   = (const float*)d_in[18];
    const float* ffn_W2   = (const float*)d_in[19];
    const float* ffn_b2   = (const float*)d_in[20];
    const float* n2g_W1   = (const float*)d_in[21];
    const float* n2g_b1   = (const float*)d_in[22];
    const float* n2g_W2   = (const float*)d_in[23];
    const float* n2g_b2   = (const float*)d_in[24];
    const float* g2n_W    = (const float*)d_in[25];
    const float* g2n_b    = (const float*)d_in[26];
    const float* dec_W1   = (const float*)d_in[27];
    const float* dec_b1   = (const float*)d_in[28];
    const float* dec_W2   = (const float*)d_in[29];
    const float* dec_b2   = (const float*)d_in[30];
    float* out = (float*)d_out;

    const int N = NN, E = EE;
    const long Etot = (long)E + N;

    // -------- workspace layout (bytes, 256-aligned) --------
    char* ws = (char*)d_ws;
    size_t off = 0;
    auto alloc = [&](size_t bytes) { size_t o = off; off += (bytes + 255) & ~(size_t)255; return o; };
    float* H    = (float*)(ws + alloc((size_t)N * 64 * 4));     // node features (f32)
    float* XL   = (float*)(ws + alloc((size_t)N * 256 * 4));    // xl / t2 / lat / dec-t
    float* XR   = (float*)(ws + alloc((size_t)N * 256 * 4));    // xr
    float* AGG  = (float*)(ws + alloc((size_t)N * 256 * 4));    // attention aggregate
    float* LOG  = (float*)(ws + alloc((size_t)Etot * 4 * 4));   // logits -> exp
    float* MX   = (float*)(ws + alloc((size_t)N * 4 * 4));
    float* DEN  = (float*)(ws + alloc((size_t)N * 4 * 4));
    bf16*  HB   = (bf16*) (ws + alloc((size_t)N * 64 * 2));     // bf16(h)
    bf16*  HB2  = (bf16*) (ws + alloc((size_t)N * 128 * 2));    // bf16 intermediates
    float* GSUM = (float*)(ws + alloc(GG * 64 * 4));
    float* GMAX = (float*)(ws + alloc(GG * 64 * 4));
    float* CNT  = (float*)(ws + alloc(GG * 4));
    float* GATE = (float*)(ws + alloc(GG * 64 * 4));
    float* EAM  = (float*)(ws + alloc(4 * 4));
    bf16*  WlP  = (bf16*) (ws + alloc((size_t)LL * 64 * 256 * 2));
    bf16*  WrP  = (bf16*) (ws + alloc((size_t)LL * 64 * 256 * 2));
    bf16*  F1P  = (bf16*) (ws + alloc((size_t)LL * 64 * 128 * 2));
    bf16*  F2P  = (bf16*) (ws + alloc((size_t)LL * 128 * 64 * 2));
    bf16*  G1P  = (bf16*) (ws + alloc((size_t)LL * 64 * 64 * 2));
    bf16*  G2P  = (bf16*) (ws + alloc((size_t)LL * 64 * 64 * 2));
    bf16*  D1P  = (bf16*) (ws + alloc((size_t)64 * 64 * 2));

    auto blk = [](long n, int b) { return (int)((n + b - 1) / b); };
    auto fill = [&](float* p, float v, long n) {
        fill_f32_kernel<<<blk(n, 256), 256, 0, stream>>>(p, v, n);
    };
    auto repack = [&](const float* B, bf16* P, int K, int Nc) {
        repack_b_kernel<<<blk((long)K * Nc, 256), 256, 0, stream>>>(B, P, K, Nc, K / 32);
    };

    // -------- per-call prep: packed bf16 weights + edge_attr mean --------
    for (int l = 0; l < LL; ++l) {
        repack(Wl     + (size_t)l * 64 * 256, WlP + (size_t)l * 64 * 256, 64, 256);
        repack(Wr     + (size_t)l * 64 * 256, WrP + (size_t)l * 64 * 256, 64, 256);
        repack(ffn_W1 + (size_t)l * 64 * 128, F1P + (size_t)l * 64 * 128, 64, 128);
        repack(ffn_W2 + (size_t)l * 128 * 64, F2P + (size_t)l * 128 * 64, 128, 64);
        repack(n2g_W1 + (size_t)l * 64 * 64,  G1P + (size_t)l * 64 * 64,  64, 64);
        repack(n2g_W2 + (size_t)l * 64 * 64,  G2P + (size_t)l * 64 * 64,  64, 64);
    }
    repack(dec_W1, D1P, 64, 64);

    fill(EAM, 0.f, 4);
    ea_sum_kernel<<<256, 256, 0, stream>>>(ea, EAM, E);
    ea_mean_kernel<<<1, 4, 0, stream>>>(EAM, 1.f / (float)E);

    // -------- encoder (writes f32 h + bf16 hb) --------
    encoder_kernel<<<blk((long)N * 64, 256), 256, 0, stream>>>(x, enc_W, enc_b, H, HB, N);

    // -------- layers --------
    const int edgeBlocks = blk(Etot, 8);
    auto gemmTiles = [&](int M, int Nc) { return blk((long)(M / 16) * (Nc / 16), 8); };
    for (int l = 0; l < LL; ++l) {
        // GATv2 source/target transforms (WMMA bf16, K=64, f32 out)
        gemm_bf16_wmma_kernel<64, 0, 0><<<gemmTiles(N, 256), 256, 0, stream>>>(
            HB, WlP + (size_t)l * 64 * 256, bl + (size_t)l * 256, XL, (bf16*)nullptr, N, 256);
        gemm_bf16_wmma_kernel<64, 0, 0><<<gemmTiles(N, 256), 256, 0, stream>>>(
            HB, WrP + (size_t)l * 64 * 256, br + (size_t)l * 256, XR, (bf16*)nullptr, N, 256);

        fill(MX, -3.0e38f, (long)N * 4);
        fill(DEN, 0.f,     (long)N * 4);
        fill(AGG, 0.f,     (long)N * 256);

        edge_logits_kernel<<<edgeBlocks, 256, 0, stream>>>(
            XL, XR, ea, ei, EAM, We + (size_t)l * 768, att_w + (size_t)l * 256,
            LOG, MX, E, N);
        edge_exp_kernel<<<blk(Etot * 4, 256), 256, 0, stream>>>(LOG, MX, DEN, ei, E, N);
        edge_agg_kernel<<<edgeBlocks, 256, 0, stream>>>(LOG, DEN, XL, ei, AGG, E, N);

        gat_ln_kernel<<<blk(N, 8), 256, 0, stream>>>(
            H, HB, AGG, gat_b + (size_t)l * 64, ln1_w + (size_t)l * 64, ln1_b + (size_t)l * 64, N);

        // FFN + LN2 (FFN1 emits bf16 directly; FFN2 K=128)
        gemm_bf16_wmma_kernel<64, 1, 1><<<gemmTiles(N, 128), 256, 0, stream>>>(
            HB, F1P + (size_t)l * 64 * 128, ffn_b1 + (size_t)l * 128,
            (float*)nullptr, HB2, N, 128);
        gemm_bf16_wmma_kernel<128, 0, 0><<<gemmTiles(N, 64), 256, 0, stream>>>(
            HB2, F2P + (size_t)l * 128 * 64, ffn_b2 + (size_t)l * 64, XL, (bf16*)nullptr, N, 64);
        add_ln_kernel<<<blk(N, 8), 256, 0, stream>>>(
            H, HB, XL, ln2_w + (size_t)l * 64, ln2_b + (size_t)l * 64, N);

        // Global context
        gemm_bf16_wmma_kernel<64, 1, 1><<<gemmTiles(N, 64), 256, 0, stream>>>(
            HB, G1P + (size_t)l * 64 * 64, n2g_b1 + (size_t)l * 64,
            (float*)nullptr, HB2, N, 64);
        gemm_bf16_wmma_kernel<64, 0, 0><<<gemmTiles(N, 64), 256, 0, stream>>>(
            HB2, G2P + (size_t)l * 64 * 64, n2g_b2 + (size_t)l * 64, XL, (bf16*)nullptr, N, 64);

        fill(GSUM, 0.f, GG * 64);
        fill(GMAX, -3.0e38f, GG * 64);
        fill(CNT, 0.f, GG);
        seg_reduce_kernel<<<blk((long)N * 64, 256), 256, 0, stream>>>(XL, batch, GSUM, GMAX, CNT, N);
        gate_kernel<<<1, 1024, 0, stream>>>(GSUM, GMAX, CNT,
                                            g2n_W + (size_t)l * 128 * 64,
                                            g2n_b + (size_t)l * 64, GATE);
        apply_gate_kernel<<<blk((long)N * 64, 256), 256, 0, stream>>>(H, HB, GATE, batch, N);
    }

    // -------- decoder --------
    gemm_bf16_wmma_kernel<64, 1, 0><<<gemmTiles(N, 64), 256, 0, stream>>>(
        HB, D1P, dec_b1, XL, (bf16*)nullptr, N, 64);
    dec2_kernel<<<blk((long)N * 2, 256), 256, 0, stream>>>(XL, dec_W2, dec_b2, out, N);
}
